// Memory_40492951666914
// MI455X (gfx1250) — matrix operand order repaired and checked by reading, not compiled, not run on hardware
//
#include <hip/hip_runtime.h>

typedef __attribute__((ext_vector_type(16))) _Float16 v16h;
typedef __attribute__((ext_vector_type(8)))  float    v8f;

#define MEM_SZ 4096
#define KD     256
#define BT     16384            // 32*512 rows
#define MROWS  32               // rows per workgroup
#define NWGS   (BT / MROWS)     // 512 workgroups
#define TOPKN  257
#define NB     192
#define HLO    0.40f
#define HHI    0.60f

#define WMMA_F16(A, B, C) \
    __builtin_amdgcn_wmma_f32_16x16x32_f16(false, (A), false, (B), (short)0, (C), false, false)

// ---------------- pack kernels: f32 -> f16 in exact WMMA register layout -------------
// A-layout (16x32 f16): lane L (0..31): row m = L&15;
//   half h: K = ((L<16)?0:8) + (h<8 ? h : h+8)
// B-layout (32x16 f16): lane L: col n = L&15; half h: K = ((L<16)?0:8) + (h<8 ? h : h+8)

// dpack: [1024 row-tiles][8 d-chunks][32 lanes][16 halves]
__global__ void pack_data_k(const float* __restrict__ data, unsigned* __restrict__ dp) {
    int o = blockIdx.x * 256 + threadIdx.x;           // dword index (2 halves)
    if (o >= BT * KD / 2) return;
    int hh = o * 2;
    int h  = hh & 15;
    int L  = (hh >> 4) & 31;
    int dc = (hh >> 9) & 7;
    int rt = hh >> 12;
    int row = rt * 16 + (L & 15);
    int K   = ((L < 16) ? 0 : 8) + ((h < 8) ? h : h + 8);
    int d   = dc * 32 + K;
    union { _Float16 f[2]; unsigned u; } p;
    p.f[0] = (_Float16)data[row * KD + d];
    p.f[1] = (_Float16)data[row * KD + d + 1];
    dp[o] = p.u;
}

// w1 (B for scores, B[d][j] = weight[j][d]): [256 j-tiles][8 d-chunks][32][16]
__global__ void pack_w1_k(const float* __restrict__ w, unsigned* __restrict__ wp) {
    int o = blockIdx.x * 256 + threadIdx.x;
    if (o >= MEM_SZ * KD / 2) return;
    int hh = o * 2;
    int h  = hh & 15;
    int L  = (hh >> 4) & 31;
    int dc = (hh >> 9) & 7;
    int jt = hh >> 12;
    int j  = jt * 16 + (L & 15);
    int d  = dc * 32 + ((L < 16) ? 0 : 8) + ((h < 8) ? h : h + 8);
    union { _Float16 f[2]; unsigned u; } p;
    p.f[0] = (_Float16)w[j * KD + d];
    p.f[1] = (_Float16)w[j * KD + d + 1];
    wp[o] = p.u;
}

// w2 (B for augment, B[j][d] = weight[j][d]): [128 j-chunks][16 d-tiles][32][16]
__global__ void pack_w2_k(const float* __restrict__ w, unsigned* __restrict__ wp) {
    int o = blockIdx.x * 256 + threadIdx.x;
    if (o >= MEM_SZ * KD / 2) return;
    int hh = o * 2;
    int h  = hh & 15;
    int L  = (hh >> 4) & 31;
    int dt = (hh >> 9) & 15;
    int jc = hh >> 13;
    int d  = dt * 16 + (L & 15);
    int j0 = jc * 32 + ((L < 16) ? 0 : 8) + ((h < 8) ? h : h + 8);
    union { _Float16 f[2]; unsigned u; } p;
    p.f[0] = (_Float16)w[(j0)     * KD + d];
    p.f[1] = (_Float16)w[(j0 + 1) * KD + d];
    wp[o] = p.u;
}

// ---------------- fused main kernel: 1 workgroup = 32 rows, 8 waves ------------------
// Wave w: row-tile rt = w>>2 (shared by GEMM1 & GEMM2), GEMM1 j-tile (w&3) of each
// 64-wide j-chunk, GEMM2 d-tiles (w&3)*4 .. (w&3)*4+3.
__global__ __launch_bounds__(256) void mem_main(const unsigned short* __restrict__ dp,
                                                const unsigned short* __restrict__ w1p,
                                                const unsigned short* __restrict__ w2p,
                                                float* __restrict__ out) {
    __shared__ float    hsum[MROWS][NB];       // 24 KB
    __shared__ unsigned hcnt[MROWS][NB];       // 24 KB
    __shared__ unsigned short att[MROWS][64];  //  4 KB sigmoid staging (row-major)

    const int tid  = threadIdx.x;
    const int wave = tid >> 5;
    const int lane = tid & 31;
    const int wg   = blockIdx.x;
    const int ncol = lane & 15;                // C-layout column / A-layout row
    const int hi16 = lane >> 4;
    const int rt   = wave >> 2;                // row-tile 0/1
    const int jtw  = wave & 3;                 // GEMM1 j-tile / GEMM2 d-group

    for (int i = tid; i < MROWS * NB; i += 256) {
        (&hsum[0][0])[i] = 0.0f;
        (&hcnt[0][0])[i] = 0u;
    }
    __syncthreads();

    // preload this wave's A row-tile (16x256 f16 = 8 v16h = 64 VGPRs)
    const v16h* Adp = ((const v16h*)dp) + (wg * 2 + rt) * 256 + lane;
    v16h a[8];
#pragma unroll
    for (int dc = 0; dc < 8; ++dc) a[dc] = Adp[dc * 32];

    v8f aug[4] = {{}, {}, {}, {}};             // 4 augment d-tiles

    for (int j0 = 0; j0 < MEM_SZ; j0 += 64) {
        // ---- GEMM1: software-pipelined B stream, prefetch distance 2 --------------
        const int jt16 = (j0 >> 4) + jtw;
        const v16h* Bw1 = ((const v16h*)w1p) + jt16 * 256 + lane;
        v16h p0 = Bw1[0 * 32];
        v16h p1 = Bw1[1 * 32];
        v8f s = {};
        s = WMMA_F16(a[0], p0, s);  p0 = Bw1[2 * 32];
        s = WMMA_F16(a[1], p1, s);  p1 = Bw1[3 * 32];
        s = WMMA_F16(a[2], p0, s);  p0 = Bw1[4 * 32];
        s = WMMA_F16(a[3], p1, s);  p1 = Bw1[5 * 32];
        s = WMMA_F16(a[4], p0, s);  p0 = Bw1[6 * 32];
        s = WMMA_F16(a[5], p1, s);  p1 = Bw1[7 * 32];
        s = WMMA_F16(a[6], p0, s);
        s = WMMA_F16(a[7], p1, s);

        __syncthreads();                       // prior GEMM2 done reading att

        // ---- sigmoid + histogram + stage (C-layout: m = e+8*hi16, n = ncol) -------
#pragma unroll
        for (int e = 0; e < 8; ++e) {
            float sc = s[e] * 0.0625f;                 // 1/sqrt(256)
            float sg = 1.0f / (1.0f + __expf(-sc));
            int m = rt * 16 + e + (hi16 << 3);
            int bin = (int)((sg - HLO) * ((float)NB / (HHI - HLO)));
            bin = bin < 0 ? 0 : (bin > NB - 1 ? NB - 1 : bin);
            atomicAdd(&hsum[m][bin], sg);
            atomicAdd(&hcnt[m][bin], 1u);
            union { _Float16 f; unsigned short u; } cv;
            cv.f = (_Float16)sg;
            att[m][jtw * 16 + ncol] = cv.u;
        }
        __syncthreads();                       // att tile complete

        // ---- GEMM2: both LDS A-gathers up front, B tiles ping-ponged --------------
        const int jcg0 = (j0 >> 5);
        const v16h* B2a = ((const v16h*)w2p) + (jcg0    ) * 512 + (jtw * 4) * 32 + lane;
        const v16h* B2b = ((const v16h*)w2p) + (jcg0 + 1) * 512 + (jtw * 4) * 32 + lane;

        union { v16h v; unsigned u[8]; } A0, A1;
        {
            const unsigned short* arow = &att[rt * 16 + ncol][0];
            const int kb = hi16 ? 8 : 0;
#pragma unroll
            for (int i = 0; i < 8; ++i) {      // A-layout gather: dword K pairs
                int off = kb + 2 * i + ((i >= 4) ? 8 : 0);
                A0.u[i] = *(const unsigned*)(arow + off);
                A1.u[i] = *(const unsigned*)(arow + 32 + off);
            }
        }
        v16h q0 = B2a[0 * 32];
        v16h q1 = B2a[1 * 32];
        aug[0] = WMMA_F16(A0.v, q0, aug[0]);  q0 = B2a[2 * 32];
        aug[1] = WMMA_F16(A0.v, q1, aug[1]);  q1 = B2a[3 * 32];
        aug[2] = WMMA_F16(A0.v, q0, aug[2]);  q0 = B2b[0 * 32];
        aug[3] = WMMA_F16(A0.v, q1, aug[3]);  q1 = B2b[1 * 32];
        aug[0] = WMMA_F16(A1.v, q0, aug[0]);  q0 = B2b[2 * 32];
        aug[1] = WMMA_F16(A1.v, q1, aug[1]);  q1 = B2b[3 * 32];
        aug[2] = WMMA_F16(A1.v, q0, aug[2]);
        aug[3] = WMMA_F16(A1.v, q1, aug[3]);
    }
    __syncthreads();

    // ---- store augment (out + 16384), C-layout scatter, coalesced across lanes ----
    float* outA = out + BT;
#pragma unroll
    for (int t = 0; t < 4; ++t) {
        const int dt = jtw * 4 + t;
#pragma unroll
        for (int e = 0; e < 8; ++e) {
            int r = wg * MROWS + rt * 16 + e + (hi16 << 3);
            outA[r * KD + dt * 16 + ncol] = aug[t][e];
        }
    }

    // ---- temporal_att: scan histogram top-down, fractional boundary bin -----------
    if (tid < MROWS) {
        float cum = 0.0f;
        unsigned cnt = 0;
        for (int b = NB - 1; b >= 0; --b) {
            unsigned c = hcnt[tid][b];
            float    v = hsum[tid][b];
            if (cnt + c >= TOPKN) {
                unsigned need = TOPKN - cnt;
                cum += v * ((float)need / (float)c);
                break;
            }
            cum += v;
            cnt += c;
        }
        out[wg * MROWS + tid] = cum * (1.0f / (float)TOPKN);
    }
}

extern "C" void kernel_launch(void* const* d_in, const int* in_sizes, int n_in,
                              void* d_out, int out_size, void* d_ws, size_t ws_size,
                              hipStream_t stream) {
    const float* data   = (const float*)d_in[0];
    const float* weight = (const float*)d_in[1];
    float* out = (float*)d_out;

    unsigned short* dpack = (unsigned short*)d_ws;      // 16384*256 halves = 8 MB
    unsigned short* w1    = dpack + (size_t)BT * KD;    // 4096*256 halves = 2 MB
    unsigned short* w2    = w1 + (size_t)MEM_SZ * KD;   // 4096*256 halves = 2 MB

    pack_data_k<<<(BT * KD / 2 + 255) / 256, 256, 0, stream>>>(data, (unsigned*)dpack);
    pack_w1_k  <<<(MEM_SZ * KD / 2 + 255) / 256, 256, 0, stream>>>(weight, (unsigned*)w1);
    pack_w2_k  <<<(MEM_SZ * KD / 2 + 255) / 256, 256, 0, stream>>>(weight, (unsigned*)w2);

    mem_main<<<NWGS, 256, 0, stream>>>(dpack, w1, w2, out);
}
